// MixtureOfExperts_46978352283681
// MI455X (gfx1250) — matrix operand order repaired and checked by reading, not compiled, not run on hardware
//
#include <hip/hip_runtime.h>

// ---------------------------------------------------------------------------
// MoE (dense expert loop) for MI455X / gfx1250, wave32 + WMMA bf16.
//   B=4, T=2048, D=1024, E=8, k=2. Tokens N = B*T = 8192.
//   out[m,n] = sum_e gates[m,e] * ( relu(x@W1_e + b1_e) @ W2_e + b2_e )
// Double-buffered LDS pipeline: one barrier per K-slab, staging of slab i+1
// overlaps the WMMAs of slab i.
// ---------------------------------------------------------------------------

#define NTOK   8192
#define DDIM   1024
#define NEXP   8

typedef __attribute__((ext_vector_type(16))) __bf16 v16bf;
typedef __attribute__((ext_vector_type(8)))  __bf16 v8bf;
typedef __attribute__((ext_vector_type(4)))  __bf16 v4bf;
typedef __attribute__((ext_vector_type(8)))  float  v8f;
typedef __attribute__((ext_vector_type(4)))  float  v4f;

union FragBF16 { v16bf v; v8bf h[2]; };

#define BM 128
#define BN 128
#define BK 32
#define LDSP 40   // row pitch in bf16 elements (80 B: 16B-aligned, skews banks)

// ---------------------------------------------------------------------------
// Gating: h = x@W_g + noise + softplus(x@W_noise); top-2 mask; softmax.
// One wave (32 lanes) per token; 8 waves per block.
// ---------------------------------------------------------------------------
__global__ __launch_bounds__(256) void moe_gates_kernel(
    const float* __restrict__ x, const float* __restrict__ noise,
    const float* __restrict__ Wg, const float* __restrict__ Wn,
    float* __restrict__ gates) {
  const int wave = threadIdx.x >> 5;
  const int lane = threadIdx.x & 31;
  const int t = blockIdx.x * 8 + wave;

  float ag[NEXP], an[NEXP];
#pragma unroll
  for (int e = 0; e < NEXP; ++e) { ag[e] = 0.f; an[e] = 0.f; }

  const float* xrow = x + (size_t)t * DDIM;
  for (int d = lane; d < DDIM; d += 32) {
    const float xv = xrow[d];
    const float* wg = Wg + (size_t)d * NEXP;
    const float* wn = Wn + (size_t)d * NEXP;
#pragma unroll
    for (int e = 0; e < NEXP; ++e) {
      ag[e] = fmaf(xv, wg[e], ag[e]);
      an[e] = fmaf(xv, wn[e], an[e]);
    }
  }
  // wave32 reduction
#pragma unroll
  for (int off = 16; off > 0; off >>= 1) {
#pragma unroll
    for (int e = 0; e < NEXP; ++e) {
      ag[e] += __shfl_xor(ag[e], off, 32);
      an[e] += __shfl_xor(an[e], off, 32);
    }
  }

  if (lane == 0) {
    float h[NEXP];
#pragma unroll
    for (int e = 0; e < NEXP; ++e) {
      const float z = an[e];
      const float sp = (z > 20.f) ? z : log1pf(expf(z));   // softplus
      h[e] = ag[e] + noise[(size_t)t * NEXP + e] + sp;
    }
    // top-2 (kth = 2nd largest; ties kept via >=)
    float m1 = -__builtin_huge_valf(), m2 = -__builtin_huge_valf();
#pragma unroll
    for (int e = 0; e < NEXP; ++e) {
      if (h[e] > m1) { m2 = m1; m1 = h[e]; }
      else if (h[e] > m2) { m2 = h[e]; }
    }
    float ex[NEXP];
    float s = 0.f;
#pragma unroll
    for (int e = 0; e < NEXP; ++e) {
      ex[e] = (h[e] >= m2) ? expf(h[e] - m1) : 0.f;
      s += ex[e];
    }
    const float inv = 1.f / s;
#pragma unroll
    for (int e = 0; e < NEXP; ++e) gates[(size_t)t * NEXP + e] = ex[e] * inv;
  }
}

// ---------------------------------------------------------------------------
// Fragment loads per CDNA5 ISA 7.12.2 (wave32):
//   A 16x32 bf16: lane m=l&15, half=l>>4 holds k runs [8h,8h+8) and [16+8h,+8)
//   B 32x16 bf16: lane n=l&15, half=l>>4 holds k runs [8h,8h+8) and [16+8h,+8)
// As: [BM][LDSP] row-major (rows=tokens); Bs: [BN][LDSP] n-major (rows=n).
// ---------------------------------------------------------------------------
__device__ __forceinline__ void load_frags(const __bf16* As, const __bf16* Bs,
                                           FragBF16 a[2], FragBF16 b[4],
                                           int wm, int wn, int lane) {
  const int half = lane >> 4;
  const int sl   = lane & 15;
#pragma unroll
  for (int mi = 0; mi < 2; ++mi) {
    const int row = wm * 32 + mi * 16 + sl;
    a[mi].h[0] = *(const v8bf*)&As[row * LDSP + half * 8];
    a[mi].h[1] = *(const v8bf*)&As[row * LDSP + 16 + half * 8];
  }
#pragma unroll
  for (int ni = 0; ni < 4; ++ni) {
    const int n = wn * 64 + ni * 16 + sl;
    b[ni].h[0] = *(const v8bf*)&Bs[n * LDSP + half * 8];
    b[ni].h[1] = *(const v8bf*)&Bs[n * LDSP + 16 + half * 8];
  }
}

__device__ __forceinline__ void wmma_compute(v8f c[2][4],
                                             const FragBF16 a[2],
                                             const FragBF16 b[4]) {
#pragma unroll
  for (int mi = 0; mi < 2; ++mi)
#pragma unroll
    for (int ni = 0; ni < 4; ++ni)
      c[mi][ni] = __builtin_amdgcn_wmma_f32_16x16x32_bf16(
          false, a[mi].v, false, b[ni].v, (short)0, c[mi][ni], false, false);
}

// Stage A tile (128x32) from f32 source, converting to bf16; ds_store_b64.
__device__ __forceinline__ void stage_a_f32(__bf16* As, const float* X,
                                            int m0, int k0, int tid) {
#pragma unroll
  for (int rep = 0; rep < 4; ++rep) {
    const int id = rep * 256 + tid;
    const int row = id >> 3, col = (id & 7) * 4;
    const v4f v = *(const v4f*)&X[(size_t)(m0 + row) * DDIM + k0 + col];
    v4bf p;
#pragma unroll
    for (int j = 0; j < 4; ++j) p[j] = (__bf16)v[j];
    *(v4bf*)&As[row * LDSP + col] = p;
  }
}

// Stage A tile (128x32) from bf16 source; ds_store_b128.
__device__ __forceinline__ void stage_a_bf16(__bf16* As, const __bf16* H,
                                             int m0, int k0, int tid) {
#pragma unroll
  for (int rep = 0; rep < 2; ++rep) {
    const int id = rep * 256 + tid;
    const int row = id >> 2, col = (id & 3) * 8;
    const v8bf v = *(const v8bf*)&H[(size_t)(m0 + row) * DDIM + k0 + col];
    *(v8bf*)&As[row * LDSP + col] = v;
  }
}

// Stage B tile (32x128) from f32 weights, bf16-converted and transposed.
__device__ __forceinline__ void stage_b_f32(__bf16* Bs, const float* W,
                                            int n0, int k0, int tid) {
#pragma unroll
  for (int rep = 0; rep < 4; ++rep) {
    const int id = rep * 256 + tid;
    const int kr = id >> 5, n4 = (id & 31) * 4;
    const v4f v = *(const v4f*)&W[(size_t)(k0 + kr) * DDIM + n0 + n4];
#pragma unroll
    for (int j = 0; j < 4; ++j) Bs[(n4 + j) * LDSP + kr] = (__bf16)v[j];
  }
}

// ---------------------------------------------------------------------------
// FFN pass 1: H = relu(X_f32 @ W_f32 + b)  -> bf16 hidden
// ---------------------------------------------------------------------------
__global__ __launch_bounds__(256) void moe_ffn1_kernel(
    const float* __restrict__ X, const float* __restrict__ W,
    const float* __restrict__ bias, __bf16* __restrict__ H) {
  __shared__ __bf16 As[2][BM * LDSP];
  __shared__ __bf16 Bs[2][BN * LDSP];

  const int m0 = blockIdx.x * BM;
  const int n0 = blockIdx.y * BN;
  const int tid = threadIdx.x;
  const int lane = tid & 31;
  const int wave = tid >> 5;
  const int wm = wave & 3, wn = wave >> 2;

  v8f c[2][4];
#pragma unroll
  for (int mi = 0; mi < 2; ++mi)
#pragma unroll
    for (int ni = 0; ni < 4; ++ni) c[mi][ni] = {};

  // prologue: fill buffer 0
  stage_a_f32(As[0], X, m0, 0, tid);
  stage_b_f32(Bs[0], W, n0, 0, tid);
  __syncthreads();

  int buf = 0;
  for (int k0 = 0; k0 < DDIM; k0 += BK) {
    FragBF16 a[2], b[4];
    load_frags(As[buf], Bs[buf], a, b, wm, wn, lane);   // ds_loads first
    if (k0 + BK < DDIM) {                               // stage next slab
      stage_a_f32(As[buf ^ 1], X, m0, k0 + BK, tid);
      stage_b_f32(Bs[buf ^ 1], W, n0, k0 + BK, tid);
    }
    if (k0 + 2 * BK < DDIM) {                           // L2 hint, 2 slabs out
      __builtin_prefetch(&X[(size_t)(m0 + (tid >> 1)) * DDIM + k0 + 2 * BK], 0, 0);
      __builtin_prefetch(&W[(size_t)(k0 + 2 * BK + (tid & 31)) * DDIM + n0], 0, 0);
    }
    wmma_compute(c, a, b);                              // overlaps staging
    __syncthreads();
    buf ^= 1;
  }

  // epilogue: +bias, relu, bf16 store
  const int half = lane >> 4, nl = lane & 15;
  float bv[4];
  int ncol[4];
#pragma unroll
  for (int ni = 0; ni < 4; ++ni) {
    ncol[ni] = n0 + wn * 64 + ni * 16 + nl;
    bv[ni] = bias[ncol[ni]];
  }
#pragma unroll
  for (int mi = 0; mi < 2; ++mi)
#pragma unroll
    for (int r = 0; r < 8; ++r) {
      const int m = m0 + wm * 32 + mi * 16 + half * 8 + r;
#pragma unroll
      for (int ni = 0; ni < 4; ++ni) {
        float v = c[mi][ni][r] + bv[ni];
        v = v > 0.f ? v : 0.f;
        H[(size_t)m * DDIM + ncol[ni]] = (__bf16)v;
      }
    }
}

// ---------------------------------------------------------------------------
// FFN pass 2: out (+)= gates[:,e] * (H_bf16 @ W_f32 + b)
// ---------------------------------------------------------------------------
__global__ __launch_bounds__(256) void moe_ffn2_kernel(
    const __bf16* __restrict__ H, const float* __restrict__ W,
    const float* __restrict__ bias, const float* __restrict__ G,
    int expert, int first, float* __restrict__ out) {
  __shared__ __bf16 As[2][BM * LDSP];
  __shared__ __bf16 Bs[2][BN * LDSP];

  const int m0 = blockIdx.x * BM;
  const int n0 = blockIdx.y * BN;
  const int tid = threadIdx.x;
  const int lane = tid & 31;
  const int wave = tid >> 5;
  const int wm = wave & 3, wn = wave >> 2;

  v8f c[2][4];
#pragma unroll
  for (int mi = 0; mi < 2; ++mi)
#pragma unroll
    for (int ni = 0; ni < 4; ++ni) c[mi][ni] = {};

  stage_a_bf16(As[0], H, m0, 0, tid);
  stage_b_f32(Bs[0], W, n0, 0, tid);
  __syncthreads();

  int buf = 0;
  for (int k0 = 0; k0 < DDIM; k0 += BK) {
    FragBF16 a[2], b[4];
    load_frags(As[buf], Bs[buf], a, b, wm, wn, lane);
    if (k0 + BK < DDIM) {
      stage_a_bf16(As[buf ^ 1], H, m0, k0 + BK, tid);
      stage_b_f32(Bs[buf ^ 1], W, n0, k0 + BK, tid);
    }
    if (k0 + 2 * BK < DDIM) {
      __builtin_prefetch(&H[(size_t)(m0 + (tid >> 1)) * DDIM + k0 + 2 * BK], 0, 0);
      __builtin_prefetch(&W[(size_t)(k0 + 2 * BK + (tid & 31)) * DDIM + n0], 0, 0);
    }
    wmma_compute(c, a, b);
    __syncthreads();
    buf ^= 1;
  }

  // epilogue: +bias, scale by per-token gate, write/accumulate
  const int half = lane >> 4, nl = lane & 15;
  float bv[4];
  int ncol[4];
#pragma unroll
  for (int ni = 0; ni < 4; ++ni) {
    ncol[ni] = n0 + wn * 64 + ni * 16 + nl;
    bv[ni] = bias[ncol[ni]];
  }
#pragma unroll
  for (int mi = 0; mi < 2; ++mi)
#pragma unroll
    for (int r = 0; r < 8; ++r) {
      const int m = m0 + wm * 32 + mi * 16 + half * 8 + r;
      const float g = G[(size_t)m * NEXP + expert];
#pragma unroll
      for (int ni = 0; ni < 4; ++ni) {
        const float v = g * (c[mi][ni][r] + bv[ni]);
        float* p = &out[(size_t)m * DDIM + ncol[ni]];
        if (first) *p = v;
        else       *p += v;
      }
    }
}

// ---------------------------------------------------------------------------
// Launcher. Inputs (setup_inputs order):
//   0:x 1:noise 2:W_g 3:W_noise 4:W1 5:b1 6:W2 7:b2 8:k
// Workspace layout: [0, 256KB) gates f32 ; [1MB, 17MB) hidden bf16
// ---------------------------------------------------------------------------
extern "C" void kernel_launch(void* const* d_in, const int* in_sizes, int n_in,
                              void* d_out, int out_size, void* d_ws, size_t ws_size,
                              hipStream_t stream) {
  (void)in_sizes; (void)n_in; (void)out_size; (void)ws_size;
  const float* x     = (const float*)d_in[0];
  const float* noise = (const float*)d_in[1];
  const float* Wg    = (const float*)d_in[2];
  const float* Wn    = (const float*)d_in[3];
  const float* W1    = (const float*)d_in[4];
  const float* b1    = (const float*)d_in[5];
  const float* W2    = (const float*)d_in[6];
  const float* b2    = (const float*)d_in[7];
  float* out = (float*)d_out;

  float*  gates  = (float*)d_ws;
  __bf16* hidden = (__bf16*)((char*)d_ws + (1u << 20));

  moe_gates_kernel<<<NTOK / 8, 256, 0, stream>>>(x, noise, Wg, Wn, gates);

  const dim3 grid(NTOK / BM, DDIM / BN);
  for (int e = 0; e < NEXP; ++e) {
    moe_ffn1_kernel<<<grid, 256, 0, stream>>>(
        x, W1 + (size_t)e * DDIM * DDIM, b1 + (size_t)e * DDIM, hidden);
    moe_ffn2_kernel<<<grid, 256, 0, stream>>>(
        hidden, W2 + (size_t)e * DDIM * DDIM, b2 + (size_t)e * DDIM,
        gates, e, (e == 0) ? 1 : 0, out);
  }
}
